// LyricSTM_54305566491411
// MI455X (gfx1250) — matrix operand order repaired and checked by reading, not compiled, not run on hardware
//
#include <hip/hip_runtime.h>

// CDNA5 (gfx1250) persistent-kernel LSTM. All big GEMMs via v_wmma_f32_16x16x32_bf16,
// fp32 accumulate, weights resident in L2, 2 device-wide barriers per timestep.
// K-loops are split at the compile-time concat boundary so the hot loops are branch-free.

typedef __attribute__((ext_vector_type(16))) __bf16 v16bf;
typedef __attribute__((ext_vector_type(8)))  float  v8f;

namespace lyric {

constexpr int T_STEPS = 1024;
constexpr int BATCH   = 128;
constexpr int HID     = 1024;
constexpr int VOCAB   = 128;
constexpr int MLPD    = 128;
constexpr int F_REAL  = 138;
constexpr int F_PAD   = 160;          // 138 padded to multiple of 32
constexpr int K1      = HID + F_PAD;  // 1184 : [h1 | x_pad]
constexpr int K2      = 2 * HID;      // 2048 : [h1_new | h2_old]
constexpr int G4      = 4 * HID;      // 4096 gate rows

constexpr int NBLOCKS  = 64;
constexpr int NTHREADS = 256;         // 8 wave32 per block -> 512 waves

// ---- workspace layout (all regions 256B aligned); total ~70.9 MB ----
constexpr size_t OFF_WC1 = 256;                               // bf16 [4096][1184]
constexpr size_t SZ_WC1  = (size_t)G4 * K1 * 2;
constexpr size_t OFF_WC2 = OFF_WC1 + SZ_WC1;                  // bf16 [4096][2048]
constexpr size_t SZ_WC2  = (size_t)G4 * K2 * 2;
constexpr size_t OFF_W1B = OFF_WC2 + SZ_WC2;                  // bf16 [128][1024]
constexpr size_t SZ_W1B  = (size_t)MLPD * HID * 2;
constexpr size_t OFF_W2B = OFF_W1B + SZ_W1B;                  // bf16 [128][128]
constexpr size_t SZ_W2B  = (size_t)VOCAB * MLPD * 2;
constexpr size_t OFF_B1  = OFF_W2B + SZ_W2B;                  // f32 [4096] (b_ih1+b_hh1)
constexpr size_t OFF_B2  = OFF_B1 + (size_t)G4 * 4;           // f32 [4096]
constexpr size_t OFF_BM1 = OFF_B2 + (size_t)G4 * 4;           // f32 [128]
constexpr size_t OFF_BM2 = OFF_BM1 + 512;                     // f32 [128]
constexpr size_t OFF_XB  = OFF_BM2 + 512;                     // bf16 [1024][128][160]
constexpr size_t SZ_XB   = (size_t)T_STEPS * BATCH * F_PAD * 2;
constexpr size_t OFF_H1B = OFF_XB + SZ_XB;                    // bf16 [2][128][1024]
constexpr size_t OFF_H2B = OFF_H1B + (size_t)2 * BATCH * HID * 2;
constexpr size_t OFF_C1  = OFF_H2B + (size_t)2 * BATCH * HID * 2;  // f32 [128][1024]
constexpr size_t OFF_C2  = OFF_C1 + (size_t)BATCH * HID * 4;

__device__ __forceinline__ unsigned short f2bf(float f) {     // RNE fp32 -> bf16
  unsigned u = __builtin_bit_cast(unsigned, f);
  u += 0x7fffu + ((u >> 16) & 1u);
  return (unsigned short)(u >> 16);
}
__device__ __forceinline__ float sigm(float x) { return 1.0f / (1.0f + __expf(-x)); }
__device__ __forceinline__ float tanh_f(float x) { return 1.0f - 2.0f / (1.0f + __expf(2.0f * x)); }

// A-operand (16x32 bf16): lane M = lane&15, g = lane>>4; K = 16g .. 16g+15 contiguous
__device__ __forceinline__ v16bf ldA(const unsigned short* p, int g) {
  union { v16bf v; float4 q[2]; } u;
  u.q[0] = *reinterpret_cast<const float4*>(p + 16 * g);
  u.q[1] = *reinterpret_cast<const float4*>(p + 16 * g + 8);
  return u.v;
}
// B-operand (32x16 bf16): lane N = lane&15, K = 16g .. 16g+15 contiguous (p pre-offset by 16g)
__device__ __forceinline__ v16bf ldB(const unsigned short* p) {
  union { v16bf v; float4 q[2]; } u;
  u.q[0] = *reinterpret_cast<const float4*>(p);
  u.q[1] = *reinterpret_cast<const float4*>(p + 8);
  return u.v;
}
__device__ __forceinline__ v8f wmma_bf16(v16bf a, v16bf b, v8f c) {
  return __builtin_amdgcn_wmma_f32_16x16x32_bf16(false, a, false, b, (short)0, c, false, false);
}

// 4-gate fused tile step: load all B fragments first, then 4 back-to-back WMMAs.
__device__ __forceinline__ void gate4_step(v8f acc[4], v16bf a,
                                           const unsigned short* b0,
                                           const unsigned short* b1,
                                           const unsigned short* b2,
                                           const unsigned short* b3) {
  v16bf f0 = ldB(b0);
  v16bf f1 = ldB(b1);
  v16bf f2 = ldB(b2);
  v16bf f3 = ldB(b3);
  acc[0] = wmma_bf16(a, f0, acc[0]);
  acc[1] = wmma_bf16(a, f1, acc[1]);
  acc[2] = wmma_bf16(a, f2, acc[2]);
  acc[3] = wmma_bf16(a, f3, acc[3]);
}

// monotonic-counter device-wide barrier (no reset races)
__device__ __forceinline__ void gbar(unsigned* cnt, unsigned target) {
  __syncthreads();
  if (threadIdx.x == 0) {
    __threadfence();
    atomicAdd(cnt, 1u);
    volatile unsigned* vc = (volatile unsigned*)cnt;
    while (*vc < target) __builtin_amdgcn_s_sleep(2);
    __threadfence();
  }
  __syncthreads();
}

} // namespace lyric

__global__ void lyric_init(unsigned* ctrl) { ctrl[0] = 0u; }

__global__ __launch_bounds__(lyric::NTHREADS, 1)
void lyric_lstm_persistent(const float* __restrict__ x,
                           const float* __restrict__ W_ih1, const float* __restrict__ b_ih1,
                           const float* __restrict__ W_hh1, const float* __restrict__ b_hh1,
                           const float* __restrict__ W_ih2, const float* __restrict__ b_ih2,
                           const float* __restrict__ W_hh2, const float* __restrict__ b_hh2,
                           const float* __restrict__ W1, const float* __restrict__ b1,
                           const float* __restrict__ W2, const float* __restrict__ b2,
                           float* __restrict__ out, char* __restrict__ ws) {
  using namespace lyric;
  unsigned*       cnt  = reinterpret_cast<unsigned*>(ws);
  unsigned short* Wc1  = reinterpret_cast<unsigned short*>(ws + OFF_WC1);
  unsigned short* Wc2  = reinterpret_cast<unsigned short*>(ws + OFF_WC2);
  unsigned short* W1b  = reinterpret_cast<unsigned short*>(ws + OFF_W1B);
  unsigned short* W2b  = reinterpret_cast<unsigned short*>(ws + OFF_W2B);
  float*          bia1 = reinterpret_cast<float*>(ws + OFF_B1);
  float*          bia2 = reinterpret_cast<float*>(ws + OFF_B2);
  float*          b1f  = reinterpret_cast<float*>(ws + OFF_BM1);
  float*          b2f  = reinterpret_cast<float*>(ws + OFF_BM2);
  unsigned short* xb   = reinterpret_cast<unsigned short*>(ws + OFF_XB);
  unsigned short* h1b  = reinterpret_cast<unsigned short*>(ws + OFF_H1B);
  unsigned short* h2b  = reinterpret_cast<unsigned short*>(ws + OFF_H2B);
  float*          c1   = reinterpret_cast<float*>(ws + OFF_C1);
  float*          c2   = reinterpret_cast<float*>(ws + OFF_C2);

  const int tid = blockIdx.x * NTHREADS + threadIdx.x;
  const int NT  = NBLOCKS * NTHREADS;

  // ---------------- prologue: bf16 weight/x staging (re-done every launch) ----------------
  #pragma unroll 1
  for (int i = tid; i < G4 * K1; i += NT) {
    int j = i / K1, k = i - j * K1;
    float v = (k < HID) ? W_hh1[j * HID + k]
            : ((k < HID + F_REAL) ? W_ih1[j * F_REAL + (k - HID)] : 0.0f);
    Wc1[i] = f2bf(v);
  }
  #pragma unroll 1
  for (int i = tid; i < G4 * K2; i += NT) {
    int j = i >> 11, k = i & (K2 - 1);
    float v = (k < HID) ? W_ih2[j * HID + k] : W_hh2[j * HID + (k - HID)];
    Wc2[i] = f2bf(v);
  }
  #pragma unroll 1
  for (int i = tid; i < MLPD * HID; i += NT)  W1b[i] = f2bf(W1[i]);
  #pragma unroll 1
  for (int i = tid; i < VOCAB * MLPD; i += NT) W2b[i] = f2bf(W2[i]);
  #pragma unroll 1
  for (int i = tid; i < G4; i += NT) { bia1[i] = b_ih1[i] + b_hh1[i]; bia2[i] = b_ih2[i] + b_hh2[i]; }
  #pragma unroll 1
  for (int i = tid; i < MLPD; i += NT) { b1f[i] = b1[i]; b2f[i] = b2[i]; }
  #pragma unroll 1
  for (int i = tid; i < T_STEPS * BATCH * F_PAD; i += NT) {
    int k = i % F_PAD, tb = i / F_PAD;
    xb[i] = (k < F_REAL) ? f2bf(x[(size_t)tb * F_REAL + k]) : (unsigned short)0;
  }
  #pragma unroll 1
  for (int i = tid; i < 2 * BATCH * HID; i += NT) { h1b[i] = 0; h2b[i] = 0; }
  #pragma unroll 1
  for (int i = tid; i < BATCH * HID; i += NT) { c1[i] = 0.0f; c2[i] = 0.0f; }

  unsigned nbar = 1;
  gbar(cnt, nbar * NBLOCKS);

  // ---------------- recurrent loop ----------------
  const int lane = threadIdx.x & 31;
  const int g = lane >> 4, q = lane & 15;
  const int wid = (blockIdx.x << 3) + (threadIdx.x >> 5);   // 0..511
  const int mt = wid >> 6, jt = wid & 63;                   // batch tile, hidden tile
  const int m0 = mt * 16, j0 = jt * 16;
  const bool is_mlp = (blockIdx.x < 8) && (threadIdx.x < 32);
  const int mm0 = blockIdx.x * 16;                          // MLP batch tile

  // per-gate B row base pointers (lane N = q), hoisted out of all loops
  const unsigned short* w1p0 = Wc1 + (size_t)(0 * HID + j0 + q) * K1 + 16 * g;
  const unsigned short* w1p1 = Wc1 + (size_t)(1 * HID + j0 + q) * K1 + 16 * g;
  const unsigned short* w1p2 = Wc1 + (size_t)(2 * HID + j0 + q) * K1 + 16 * g;
  const unsigned short* w1p3 = Wc1 + (size_t)(3 * HID + j0 + q) * K1 + 16 * g;
  const unsigned short* w2p0 = Wc2 + (size_t)(0 * HID + j0 + q) * K2 + 16 * g;
  const unsigned short* w2p1 = Wc2 + (size_t)(1 * HID + j0 + q) * K2 + 16 * g;
  const unsigned short* w2p2 = Wc2 + (size_t)(2 * HID + j0 + q) * K2 + 16 * g;
  const unsigned short* w2p3 = Wc2 + (size_t)(3 * HID + j0 + q) * K2 + 16 * g;

  __shared__ __align__(16) unsigned short smm[16 * MLPD];   // SELU output tile (wave-local)

  for (int t = 0; t < T_STEPS; ++t) {
    const int p = t & 1;
    const unsigned short* h1_old = h1b + (size_t)p * BATCH * HID;
    unsigned short*       h1_new = h1b + (size_t)(p ^ 1) * BATCH * HID;
    const unsigned short* h2_old = h2b + (size_t)p * BATCH * HID;
    unsigned short*       h2_new = h2b + (size_t)(p ^ 1) * BATCH * HID;

    // ---- Phase 1: gates1 = bias + [h1_old | x_t] @ Wc1^T, then cell1 elementwise ----
    {
      v8f acc[4];
      #pragma unroll
      for (int gt = 0; gt < 4; ++gt) {
        float bv = bia1[gt * HID + j0 + q];
        #pragma unroll
        for (int r = 0; r < 8; ++r) acc[gt][r] = bv;
      }
      const unsigned short* hrow = h1_old + (size_t)(m0 + q) * HID;
      const unsigned short* xrow = xb + ((size_t)t * BATCH + m0 + q) * F_PAD;
      #pragma unroll 2
      for (int k0 = 0; k0 < HID; k0 += 32) {                // h-region (branch-free)
        v16bf a = ldA(hrow + k0, g);
        gate4_step(acc, a, w1p0 + k0, w1p1 + k0, w1p2 + k0, w1p3 + k0);
      }
      #pragma unroll 1
      for (int k0 = 0; k0 < F_PAD; k0 += 32) {              // x-region (branch-free)
        v16bf a = ldA(xrow + k0, g);
        gate4_step(acc, a, w1p0 + HID + k0, w1p1 + HID + k0,
                           w1p2 + HID + k0, w1p3 + HID + k0);
      }
      #pragma unroll
      for (int r = 0; r < 8; ++r) {
        int brow = m0 + r + 8 * g, j = j0 + q;
        float iv = sigm(acc[0][r]);
        float fv = sigm(acc[1][r]);
        float gv = tanh_f(acc[2][r]);
        float ov = sigm(acc[3][r]);
        float cc = fv * c1[(size_t)brow * HID + j] + iv * gv;
        c1[(size_t)brow * HID + j] = cc;
        h1_new[(size_t)brow * HID + j] = f2bf(ov * tanh_f(cc));
      }
    }
    ++nbar; gbar(cnt, nbar * NBLOCKS);

    // ---- Phase 2: gates2 = bias + [h1_new | h2_old] @ Wc2^T, then cell2 elementwise ----
    {
      v8f acc[4];
      #pragma unroll
      for (int gt = 0; gt < 4; ++gt) {
        float bv = bia2[gt * HID + j0 + q];
        #pragma unroll
        for (int r = 0; r < 8; ++r) acc[gt][r] = bv;
      }
      const unsigned short* hrowA = h1_new + (size_t)(m0 + q) * HID;
      const unsigned short* hrowB = h2_old + (size_t)(m0 + q) * HID;
      #pragma unroll 2
      for (int k0 = 0; k0 < HID; k0 += 32) {                // h1_new region
        v16bf a = ldA(hrowA + k0, g);
        gate4_step(acc, a, w2p0 + k0, w2p1 + k0, w2p2 + k0, w2p3 + k0);
      }
      #pragma unroll 2
      for (int k0 = 0; k0 < HID; k0 += 32) {                // h2_old region
        v16bf a = ldA(hrowB + k0, g);
        gate4_step(acc, a, w2p0 + HID + k0, w2p1 + HID + k0,
                           w2p2 + HID + k0, w2p3 + HID + k0);
      }
      #pragma unroll
      for (int r = 0; r < 8; ++r) {
        int brow = m0 + r + 8 * g, j = j0 + q;
        float iv = sigm(acc[0][r]);
        float fv = sigm(acc[1][r]);
        float gv = tanh_f(acc[2][r]);
        float ov = sigm(acc[3][r]);
        float cc = fv * c2[(size_t)brow * HID + j] + iv * gv;
        c2[(size_t)brow * HID + j] = cc;
        h2_new[(size_t)brow * HID + j] = f2bf(ov * tanh_f(cc));
      }
    }
    ++nbar; gbar(cnt, nbar * NBLOCKS);

    // ---- Phase 3 (8 waves, overlaps next step's Phase 1 of other waves):
    //      out[t] = SELU(h2_new @ W1^T + b1) @ W2^T + b2 ----
    if (is_mlp) {
      const unsigned short* hrow = h2_new + (size_t)(mm0 + q) * HID;
      #pragma unroll 1
      for (int nt = 0; nt < 8; ++nt) {           // mlp1 + SELU -> LDS
        v8f acc;
        float bv = b1f[nt * 16 + q];
        #pragma unroll
        for (int r = 0; r < 8; ++r) acc[r] = bv;
        const unsigned short* wp = W1b + (size_t)(nt * 16 + q) * HID + 16 * g;
        #pragma unroll 2
        for (int k0 = 0; k0 < HID; k0 += 32) {
          v16bf a = ldA(hrow + k0, g);
          v16bf b = ldB(wp + k0);
          acc = wmma_bf16(a, b, acc);
        }
        #pragma unroll
        for (int r = 0; r < 8; ++r) {
          float v = acc[r];
          v = 1.0507009873554805f * (v > 0.0f ? v : 1.6732632423543772f * (__expf(v) - 1.0f));
          smm[(r + 8 * g) * MLPD + nt * 16 + q] = f2bf(v);
        }
      }
      #pragma unroll 1
      for (int vt = 0; vt < 8; ++vt) {           // mlp2 -> out
        v8f acc;
        float bv = b2f[vt * 16 + q];
        #pragma unroll
        for (int r = 0; r < 8; ++r) acc[r] = bv;
        const unsigned short* wp = W2b + (size_t)(vt * 16 + q) * MLPD + 16 * g;
        #pragma unroll
        for (int k0 = 0; k0 < MLPD; k0 += 32) {
          v16bf a = ldA(smm + q * MLPD + k0, g);
          v16bf b = ldB(wp + k0);
          acc = wmma_bf16(a, b, acc);
        }
        #pragma unroll
        for (int r = 0; r < 8; ++r)
          out[((size_t)t * BATCH + mm0 + r + 8 * g) * VOCAB + vt * 16 + q] = acc[r];
      }
    }
    // no barrier: Phase 3 reads only h2_new/W*/LDS; next Phase 1 touches h1b[p]/c1 only,
    // and the next barrier (inside the t+1 loop) orders Phase 3 before Phase 2(t+1).
  }
}

extern "C" void kernel_launch(void* const* d_in, const int* in_sizes, int n_in,
                              void* d_out, int out_size, void* d_ws, size_t ws_size,
                              hipStream_t stream) {
  (void)in_sizes; (void)n_in; (void)out_size; (void)ws_size;
  const float* x     = (const float*)d_in[0];
  const float* W_ih1 = (const float*)d_in[1];
  const float* b_ih1 = (const float*)d_in[2];
  const float* W_hh1 = (const float*)d_in[3];
  const float* b_hh1 = (const float*)d_in[4];
  const float* W_ih2 = (const float*)d_in[5];
  const float* b_ih2 = (const float*)d_in[6];
  const float* W_hh2 = (const float*)d_in[7];
  const float* b_hh2 = (const float*)d_in[8];
  const float* W1    = (const float*)d_in[9];
  const float* b1    = (const float*)d_in[10];
  const float* W2    = (const float*)d_in[11];
  const float* b2    = (const float*)d_in[12];
  float* out = (float*)d_out;
  char*  ws  = (char*)d_ws;   // requires ~71 MB scratch

  lyric_init<<<1, 1, 0, stream>>>((unsigned*)ws);
  lyric_lstm_persistent<<<lyric::NBLOCKS, lyric::NTHREADS, 0, stream>>>(
      x, W_ih1, b_ih1, W_hh1, b_hh1, W_ih2, b_ih2, W_hh2, b_hh2,
      W1, b1, W2, b2, out, ws);
}
